// TransformerProModel_60799557042411
// MI455X (gfx1250) — compile-verified, matrix operand order checked by reading, not run on hardware
//
#include <hip/hip_runtime.h>
#include <math.h>

// Problem constants (reference: B=4, S=4096, D=256, H=1024, E=8, K=2)
constexpr int NTOK = 4 * 4096;   // 16384 tokens
constexpr int DIM  = 256;
constexpr int HID  = 1024;
constexpr int NEXP = 8;
constexpr int MTILE = 64;                  // rows per block (4 WMMA M-tiles)
constexpr int MAXTILES = NTOK / MTILE;     // 256 tiles per expert worst case
constexpr int NTHREADS = 512;              // 16 waves

typedef __bf16 bf16;
typedef __attribute__((ext_vector_type(8)))  __bf16 bf16x8;
typedef __attribute__((ext_vector_type(16))) __bf16 v16bf;
typedef __attribute__((ext_vector_type(8)))  float  v8f;

__device__ __forceinline__ v16bf mk16(bf16x8 lo, bf16x8 hi) {
  v16bf r;
#pragma unroll
  for (int i = 0; i < 8; ++i) { r[i] = lo[i]; r[i + 8] = hi[i]; }
  return r;
}

__device__ __forceinline__ float gelu_exact(float v) {
  return 0.5f * v * (1.0f + erff(v * 0.70710678118654752f));
}

// ---------------- prepass: fp32 -> bf16 conversions (+ transposes) ----------
__global__ void cvt_x_kernel(const float* __restrict__ x, bf16* __restrict__ xb) {
  int i = blockIdx.x * blockDim.x + threadIdx.x;      // over NTOK*DIM
  xb[i] = (bf16)x[i];
}

// w1: [E, D, H] -> w1p: [E, H, D]  (B^T, so B fragments are K-contiguous)
__global__ void cvt_w1_kernel(const float* __restrict__ w1, bf16* __restrict__ w1p) {
  size_t i = (size_t)blockIdx.x * blockDim.x + threadIdx.x;  // over E*D*H
  size_t e = i / ((size_t)DIM * HID);
  size_t r = i % ((size_t)DIM * HID);
  size_t h = r / DIM;
  size_t d = r % DIM;
  w1p[i] = (bf16)w1[e * DIM * HID + d * HID + h];
}

// w2: [E, H, D] -> w2p: [E, D, H]
__global__ void cvt_w2_kernel(const float* __restrict__ w2, bf16* __restrict__ w2p) {
  size_t i = (size_t)blockIdx.x * blockDim.x + threadIdx.x;  // over E*H*D
  size_t e = i / ((size_t)DIM * HID);
  size_t r = i % ((size_t)DIM * HID);
  size_t d = r / HID;
  size_t h = r % HID;
  w2p[i] = (bf16)w2[e * HID * DIM + h * DIM + d];
}

// ---------------- router: softmax, top-2, renorm, expert-list append --------
__global__ void router_kernel(const float* __restrict__ x,
                              const float* __restrict__ wr,
                              const float* __restrict__ br,
                              float* __restrict__ probs_out,
                              int* __restrict__ cnt,
                              int* __restrict__ tokList,
                              int* __restrict__ slotList,
                              float* __restrict__ wList) {
  __shared__ float swr[DIM * NEXP];  // 8KB, broadcast weights
  int tid = threadIdx.x;
  for (int i = tid; i < DIM * NEXP; i += blockDim.x) swr[i] = wr[i];
  __syncthreads();

  int t = blockIdx.x * blockDim.x + tid;
  float acc[NEXP];
#pragma unroll
  for (int e = 0; e < NEXP; ++e) acc[e] = br[e];

  const float4* xr = (const float4*)(x + (size_t)t * DIM);
#pragma unroll 4
  for (int d4 = 0; d4 < DIM / 4; ++d4) {
    float4 xv = xr[d4];
    float xs[4] = {xv.x, xv.y, xv.z, xv.w};
#pragma unroll
    for (int j = 0; j < 4; ++j) {
      int d = d4 * 4 + j;
#pragma unroll
      for (int e = 0; e < NEXP; ++e) acc[e] += xs[j] * swr[d * NEXP + e];
    }
  }

  // softmax
  float m = acc[0];
#pragma unroll
  for (int e = 1; e < NEXP; ++e) m = fmaxf(m, acc[e]);
  float p[NEXP], s = 0.f;
#pragma unroll
  for (int e = 0; e < NEXP; ++e) { p[e] = __expf(acc[e] - m); s += p[e]; }
  float inv = 1.0f / s;
#pragma unroll
  for (int e = 0; e < NEXP; ++e) {
    p[e] *= inv;
    probs_out[(size_t)t * NEXP + e] = p[e];
  }

  // top-2 (ties -> lowest index, matching lax.top_k)
  int i0 = 0; float p0 = p[0];
#pragma unroll
  for (int e = 1; e < NEXP; ++e) if (p[e] > p0) { p0 = p[e]; i0 = e; }
  int i1 = (i0 == 0) ? 1 : 0; float p1 = p[i1];
#pragma unroll
  for (int e = 0; e < NEXP; ++e)
    if (e != i0 && p[e] > p1) { p1 = p[e]; i1 = e; }

  float rn = 1.0f / (p0 + p1);
  float w0 = p0 * rn, w1 = p1 * rn;

  int pos0 = atomicAdd(&cnt[i0], 1);
  tokList[i0 * NTOK + pos0] = t; slotList[i0 * NTOK + pos0] = 0; wList[i0 * NTOK + pos0] = w0;
  int pos1 = atomicAdd(&cnt[i1], 1);
  tokList[i1 * NTOK + pos1] = t; slotList[i1 * NTOK + pos1] = 1; wList[i1 * NTOK + pos1] = w1;
}

// ---------------- fused expert FFN: gathered 64-row tile, bf16 WMMA ---------
// Each B-fragment is reused by 4 WMMAs (four M-tiles) -> 1/4 the weight traffic.
__global__ void __launch_bounds__(NTHREADS)
moe_gemm_kernel(const bf16* __restrict__ xb,
                const bf16* __restrict__ w1p,   // [E][H][D]
                const bf16* __restrict__ w2p,   // [E][D][H]
                const float* __restrict__ b1,   // [E][H]
                const float* __restrict__ b2,   // [E][D]
                const int* __restrict__ cnt,
                const int* __restrict__ tokList,
                const int* __restrict__ slotList,
                const float* __restrict__ wList,
                float* __restrict__ yslot) {    // [NTOK][2][D]
  const int e = blockIdx.y;
  const int tile = blockIdx.x;
  const int nAvail = cnt[e] - tile * MTILE;
  if (nAvail <= 0) return;
  const int rows = nAvail < MTILE ? nAvail : MTILE;
  const int base = e * NTOK + tile * MTILE;

  // 64 x 1024 bf16 hidden tile = 128KB (gfx1250 allows up to 320KB/WG).
  // 8-element groups XOR-swizzled by (row&7) to spread LDS banks.
  __shared__ __align__(16) bf16 Hs[MTILE][HID];

  const int tid  = threadIdx.x;
  const int wv   = tid >> 5;          // wave 0..15
  const int lane = tid & 31;
  const int mrow = lane & 15;         // A row within M-tile / B column within 16
  const int half = lane >> 4;         // K-split select
  const int koff = half ? 8 : 0;

  // gather the tokens backing this lane's four A rows (invalid rows -> row 0)
  int tokA[4];
#pragma unroll
  for (int mt = 0; mt < 4; ++mt) {
    int r = mt * 16 + mrow;
    tokA[mt] = tokList[base + (r < rows ? r : 0)];
  }

  // ---------------- GEMM1: (64xD) @ w1 -> (64xH), wave owns 64 H-columns --
  v8f acc[4][4];
#pragma unroll
  for (int mt = 0; mt < 4; ++mt)
#pragma unroll
    for (int ct = 0; ct < 4; ++ct)
#pragma unroll
      for (int r = 0; r < 8; ++r) acc[mt][ct][r] = 0.0f;

  const bf16* B1 = w1p + (size_t)e * HID * DIM;   // [h][d]
#pragma unroll
  for (int kt = 0; kt < DIM / 32; ++kt) {
    const int kb = kt * 32;
    v16bf Af[4];
#pragma unroll
    for (int mt = 0; mt < 4; ++mt) {
      const bf16* ap = xb + (size_t)tokA[mt] * DIM + kb + koff;
      Af[mt] = mk16(*(const bf16x8*)ap, *(const bf16x8*)(ap + 16));
    }
#pragma unroll
    for (int ct = 0; ct < 4; ++ct) {
      const int col = wv * 64 + ct * 16 + mrow;
      const bf16* bp = B1 + (size_t)col * DIM + kb + koff;
      if (kt + 1 < DIM / 32) __builtin_prefetch(bp + 32, 0, 1);  // next K tile
      v16bf Bf = mk16(*(const bf16x8*)bp, *(const bf16x8*)(bp + 16));
#pragma unroll
      for (int mt = 0; mt < 4; ++mt)
        acc[mt][ct] = __builtin_amdgcn_wmma_f32_16x16x32_bf16(
            false, Af[mt], false, Bf, (short)0, acc[mt][ct], false, false);
    }
  }

  // bias + exact GELU -> Hs (bf16, swizzled)
#pragma unroll
  for (int ct = 0; ct < 4; ++ct) {
    const int col = wv * 64 + ct * 16 + mrow;
    const float bb = b1[e * HID + col];
    const int g = col >> 3, cl = col & 7;
#pragma unroll
    for (int mt = 0; mt < 4; ++mt)
#pragma unroll
      for (int r = 0; r < 8; ++r) {
        const int mm = r + 8 * half + 16 * mt;   // C/D layout rows
        const int sw = (((g ^ (mm & 7)) << 3) | cl);
        Hs[mm][sw] = (bf16)gelu_exact(acc[mt][ct][r] + bb);
      }
  }
  __syncthreads();

  // ---------------- GEMM2: (64xH) @ w2 -> (64xD), wave owns 16 D-columns ---
  v8f acc2[4];
#pragma unroll
  for (int mt = 0; mt < 4; ++mt)
#pragma unroll
    for (int r = 0; r < 8; ++r) acc2[mt][r] = 0.0f;

  const int col2 = wv * 16 + mrow;                // this wave's output column
  const bf16* B2 = w2p + (size_t)e * DIM * HID;   // [d][h]
  for (int kt = 0; kt < HID / 32; ++kt) {
    const int kb = kt * 32;
    const int g0 = (kb + koff) >> 3, g1 = (kb + 16 + koff) >> 3;
    v16bf Af[4];
#pragma unroll
    for (int mt = 0; mt < 4; ++mt) {
      const int row = mt * 16 + mrow;
      const int s = row & 7;
      bf16x8 a0 = *(const bf16x8*)(&Hs[row][(g0 ^ s) << 3]);
      bf16x8 a1 = *(const bf16x8*)(&Hs[row][(g1 ^ s) << 3]);
      Af[mt] = mk16(a0, a1);
    }
    const bf16* bp = B2 + (size_t)col2 * HID + kb + koff;
    if (kt + 1 < HID / 32) __builtin_prefetch(bp + 32, 0, 1);
    v16bf Bf = mk16(*(const bf16x8*)bp, *(const bf16x8*)(bp + 16));
#pragma unroll
    for (int mt = 0; mt < 4; ++mt)
      acc2[mt] = __builtin_amdgcn_wmma_f32_16x16x32_bf16(
          false, Af[mt], false, Bf, (short)0, acc2[mt], false, false);
  }

  // bias + combine weight, scatter into per-(token,slot) buffer
  const float bb2 = b2[e * DIM + col2];
#pragma unroll
  for (int mt = 0; mt < 4; ++mt)
#pragma unroll
    for (int r = 0; r < 8; ++r) {
      const int mm = r + 8 * half + 16 * mt;
      if (mm < rows) {
        const int   tok = tokList[base + mm];
        const int   sl  = slotList[base + mm];
        const float wt  = wList[base + mm];
        const float y   = acc2[mt][r] + bb2;
        yslot[((size_t)tok * 2 + sl) * DIM + col2] = wt * y;
      }
    }
}

// ---------------- final reduce: out[t] = slot0[t] + slot1[t] ----------------
__global__ void reduce_kernel(const float* __restrict__ yslot, float* __restrict__ out) {
  size_t i = (size_t)blockIdx.x * blockDim.x + threadIdx.x;  // over NTOK*DIM/4
  const size_t t = i / (DIM / 4), d = i % (DIM / 4);
  const float4* a = (const float4*)yslot;
  float4 v0 = a[(t * 2) * (DIM / 4) + d];
  float4 v1 = a[(t * 2 + 1) * (DIM / 4) + d];
  float4 r; r.x = v0.x + v1.x; r.y = v0.y + v1.y; r.z = v0.z + v1.z; r.w = v0.w + v1.w;
  ((float4*)out)[i] = r;
}

extern "C" void kernel_launch(void* const* d_in, const int* in_sizes, int n_in,
                              void* d_out, int out_size, void* d_ws, size_t ws_size,
                              hipStream_t stream) {
  const float* x   = (const float*)d_in[0];
  const float* wr  = (const float*)d_in[1];
  const float* br  = (const float*)d_in[2];
  const float* w1  = (const float*)d_in[3];
  const float* b1  = (const float*)d_in[4];
  const float* w2  = (const float*)d_in[5];
  const float* b2  = (const float*)d_in[6];
  float* out       = (float*)d_out;                         // [NTOK, DIM]
  float* probs_out = out + (size_t)NTOK * DIM;              // [NTOK, NEXP]

  // workspace carve-up (all sub-buffers 16B aligned)
  char* ws = (char*)d_ws;
  int*   cnt      = (int*)ws;                               // NEXP (256B slab)
  int*   tokList  = (int*)(ws + 256);                       // NEXP*NTOK
  int*   slotList = tokList + (size_t)NEXP * NTOK;
  float* wList    = (float*)(slotList + (size_t)NEXP * NTOK);
  bf16*  xb       = (bf16*)(wList + (size_t)NEXP * NTOK);   // NTOK*DIM
  bf16*  w1p      = xb + (size_t)NTOK * DIM;                // NEXP*HID*DIM
  bf16*  w2p      = w1p + (size_t)NEXP * DIM * HID;         // NEXP*DIM*HID
  float* yslot    = (float*)(w2p + (size_t)NEXP * DIM * HID); // NTOK*2*DIM

  hipMemsetAsync(cnt, 0, 256, stream);

  cvt_x_kernel<<<(NTOK * DIM) / 256, 256, 0, stream>>>(x, xb);
  cvt_w1_kernel<<<(NEXP * DIM * HID) / 256, 256, 0, stream>>>(w1, w1p);
  cvt_w2_kernel<<<(NEXP * DIM * HID) / 256, 256, 0, stream>>>(w2, w2p);

  router_kernel<<<NTOK / 256, 256, 0, stream>>>(x, wr, br, probs_out,
                                                cnt, tokList, slotList, wList);

  moe_gemm_kernel<<<dim3(MAXTILES, NEXP), NTHREADS, 0, stream>>>(
      xb, w1p, w2p, b1, b2, cnt, tokList, slotList, wList, yslot);

  reduce_kernel<<<(NTOK * DIM / 4) / 256, 256, 0, stream>>>(yslot, out);
}
